// MultiHeadSelfAttention_4252017623092
// MI455X (gfx1250) — matrix operand order repaired and checked by reading, not compiled, not run on hardware
//
#include <hip/hip_runtime.h>
#include <stdint.h>

typedef _Float16 f16;
typedef __attribute__((ext_vector_type(16))) _Float16 v16h;
typedef __attribute__((ext_vector_type(8)))  float    v8f;

#define Bb 2
#define Tt 2048
#define Cc 1024
#define Hh 16
#define Dd 64
#define Mm (Bb*Tt)        // 4096 rows (b,t)
#define NQKV (3*Cc)       // 3072

// ---------------------------------------------------------------------------
// Fragment loaders (wave32, CDNA5 ISA 7.12.2 layouts). Generic pointers:
// with __shared__ bases these lower to ds_load_b128 after inlining.
__device__ __forceinline__ v16h load_a_frag(const f16* base, int ld) {
  const int lane = threadIdx.x & 31;
  const int sel  = lane >> 4;
  const int row  = lane & 15;
  const f16* p = base + (size_t)row * ld + sel * 8;
  union { float4 f[2]; v16h v; } u;
  u.f[0] = *(const float4*)(p);        // k = 8*sel + 0..7
  u.f[1] = *(const float4*)(p + 16);   // k = 8*sel + 16..23
  return u.v;
}

__device__ __forceinline__ v16h load_b_frag(const f16* base, int ld) {
  const int lane = threadIdx.x & 31;
  const int sel  = lane >> 4;
  const int n    = lane & 15;
  const f16* p = base + (size_t)n * ld + sel * 8;
  union { float4 f[2]; v16h v; } u;
  u.f[0] = *(const float4*)(p);
  u.f[1] = *(const float4*)(p + 16);
  return u.v;
}

__device__ __forceinline__ v8f wmma16(v16h a, v16h b, v8f c) {
  return __builtin_amdgcn_wmma_f32_16x16x32_f16(false, a, false, b, (short)0, c,
                                                false, false);
}

// ---------------------------------------------------------------------------
// CDNA5 async global->LDS copy (ASYNCcnt), 16 bytes per lane per op.
__device__ __forceinline__ void async_copy_b128(uint32_t lds_off, const void* gaddr) {
  asm volatile("global_load_async_to_lds_b128 %0, %1, off"
               :: "v"(lds_off), "v"(gaddr)
               : "memory");
}
__device__ __forceinline__ void wait_async0() {
  asm volatile("s_wait_asynccnt 0x0" ::: "memory");
}

// Stage a row-major [ROWS][8<<SEGSHIFT f16] tile into LDS (same geometry,
// densely packed). 256 threads; fully unrolled (CHUNKS/256 iterations).
template <int SEGSHIFT, int CHUNKS>
__device__ __forceinline__ void stage_tile(uint32_t lds_base, const f16* g,
                                           int ldg) {
#pragma unroll
  for (int i = 0; i < CHUNKS / 256; ++i) {
    int c = (int)threadIdx.x + i * 256;
    int row = c >> SEGSHIFT, seg = c & ((1 << SEGSHIFT) - 1);
    uint32_t lds_off = lds_base + (uint32_t)(((row << SEGSHIFT) + seg) * 16);
    async_copy_b128(lds_off, (const void*)(g + (size_t)row * ldg + seg * 8));
  }
}

// ---------------------------------------------------------------------------
// fp32 -> f16 convert / transpose-convert
__global__ void k_cvt(const float* __restrict__ src, f16* __restrict__ dst, int n) {
  int i = blockIdx.x * blockDim.x + threadIdx.x;
  if (i < n) dst[i] = (f16)src[i];
}
// dst[n*K + k] = src[k*N + n]  (weights N-major for B-fragments)
__global__ void k_cvt_t(const float* __restrict__ src, f16* __restrict__ dst,
                        int K, int N) {
  int i = blockIdx.x * blockDim.x + threadIdx.x;
  if (i < K * N) {
    int n = i / K, k = i - n * K;
    dst[i] = (f16)src[(size_t)k * N + n];
  }
}

// ---------------------------------------------------------------------------
// LDS-staged 128x128 GEMM core, K = 1024, staged in 64-deep double-buffered
// chunks (16 WMMA per barrier). As/Bs: [2][128][64] f16 (32 KB each).
// Per k-step: all 6 fragments loaded up front, then 8 back-to-back WMMAs.
__device__ __forceinline__ void gemm_core(const f16* Ag, const f16* Bg,
                                          f16 (*As)[128][64], f16 (*Bs)[128][64],
                                          int wm, int wn, v8f acc[2][4]) {
  const uint32_t aB = (uint32_t)(uintptr_t)&As[0][0][0];
  const uint32_t bB = (uint32_t)(uintptr_t)&Bs[0][0][0];
  stage_tile<3, 1024>(aB, Ag, Cc);      // 128 x 64 tile, k-chunk 0
  stage_tile<3, 1024>(bB, Bg, Cc);
  wait_async0();
  __syncthreads();
  for (int k0 = 0; k0 < Cc; k0 += 64) {
    const int buf = (k0 >> 6) & 1;
    if (k0 + 64 < Cc) {   // stage next chunk while computing this one
      stage_tile<3, 1024>(aB + (uint32_t)(buf ^ 1) * 16384u, Ag + k0 + 64, Cc);
      stage_tile<3, 1024>(bB + (uint32_t)(buf ^ 1) * 16384u, Bg + k0 + 64, Cc);
    }
    const f16* Asp = &As[buf][wm * 32][0];
    const f16* Bsp = &Bs[buf][wn * 64][0];
#pragma unroll
    for (int kk = 0; kk < 64; kk += 32) {
      // gather all fragments first -> one clause of ds_load_b128
      v16h a0 = load_a_frag(Asp + kk, 64);
      v16h a1 = load_a_frag(Asp + 16 * 64 + kk, 64);
      v16h b0 = load_b_frag(Bsp + 0 * 16 * 64 + kk, 64);
      v16h b1 = load_b_frag(Bsp + 1 * 16 * 64 + kk, 64);
      v16h b2 = load_b_frag(Bsp + 2 * 16 * 64 + kk, 64);
      v16h b3 = load_b_frag(Bsp + 3 * 16 * 64 + kk, 64);
      // 8 back-to-back WMMAs
      acc[0][0] = wmma16(a0, b0, acc[0][0]);
      acc[1][0] = wmma16(a1, b0, acc[1][0]);
      acc[0][1] = wmma16(a0, b1, acc[0][1]);
      acc[1][1] = wmma16(a1, b1, acc[1][1]);
      acc[0][2] = wmma16(a0, b2, acc[0][2]);
      acc[1][2] = wmma16(a1, b2, acc[1][2]);
      acc[0][3] = wmma16(a0, b3, acc[0][3]);
      acc[1][3] = wmma16(a1, b3, acc[1][3]);
    }
    wait_async0();       // my async stores into buf^1 are done
    __syncthreads();     // everyone finished reading buf / staging buf^1
  }
}

// ---------------------------------------------------------------------------
// QKV GEMM: [4096 x 1024] * [1024 x 3072] -> Q[B,H,T,D], K[B,H,T,D],
// V^T[B,H,D,T]  (all f16)
__global__ __launch_bounds__(256) void k_qkv(const f16* __restrict__ X,
                                             const f16* __restrict__ WaT,
                                             f16* __restrict__ Q,
                                             f16* __restrict__ Kd,
                                             f16* __restrict__ Vt) {
  __shared__ f16 As[2][128][64];
  __shared__ f16 Bs[2][128][64];
  const int wid  = threadIdx.x >> 5;
  const int lane = threadIdx.x & 31;
  const int sel  = lane >> 4;
  const int wm = wid & 3, wn = wid >> 2;
  const int m0 = blockIdx.y * 128 + wm * 32;
  const int n0 = blockIdx.x * 128 + wn * 64;

  v8f acc[2][4] = {};
  gemm_core(X + (size_t)(blockIdx.y * 128) * Cc,
            WaT + (size_t)(blockIdx.x * 128) * Cc, As, Bs, wm, wn, acc);

#pragma unroll
  for (int mt = 0; mt < 2; ++mt)
#pragma unroll
    for (int nt = 0; nt < 4; ++nt)
#pragma unroll
      for (int r = 0; r < 8; ++r) {
        int row = m0 + mt * 16 + r + 8 * sel;     // (b,t)
        int col = n0 + nt * 16 + (lane & 15);     // 0..3071
        f16 v = (f16)acc[mt][nt][r];
        int b = row >> 11, t = row & (Tt - 1);
        if (col < Cc) {
          int h = col >> 6, d = col & 63;
          Q[(((size_t)b * Hh + h) * Tt + t) * Dd + d] = v;
        } else if (col < 2 * Cc) {
          int c = col - Cc; int h = c >> 6, d = c & 63;
          Kd[(((size_t)b * Hh + h) * Tt + t) * Dd + d] = v;
        } else {
          int c = col - 2 * Cc; int h = c >> 6, d = c & 63;
          Vt[(((size_t)b * Hh + h) * Dd + d) * Tt + t] = v;
        }
      }
}

// ---------------------------------------------------------------------------
// Flash attention, LDS-staged 64-key chunks: one block = one (b,h) x 8
// q-tiles. Grid 512 = 32 heads x 16 q-groups.
__global__ __launch_bounds__(256) void k_attn(const f16* __restrict__ Q,
                                              const f16* __restrict__ Kd,
                                              const f16* __restrict__ Vt,
                                              f16* __restrict__ Y) {
  __shared__ f16 Ks[2][64][64];     // [t][d] chunk, 8 KB each
  __shared__ f16 Vs[2][64][64];     // [d][t] chunk, 8 KB each
  __shared__ f16 pbuf[8][16][64];   // per-wave 16x64 prob tile
  const int wid  = threadIdx.x >> 5;
  const int lane = threadIdx.x & 31;
  const int sel  = lane >> 4;
  const int bh   = blockIdx.x >> 4;          // 0..31
  const int qblk = blockIdx.x & 15;
  const int b = bh >> 4, h = bh & 15;
  const int q0 = (qblk * 8 + wid) * 16;

  const f16* Qp = Q  + (size_t)bh * Tt * Dd;
  const f16* Kp = Kd + (size_t)bh * Tt * Dd;
  const f16* Vp = Vt + (size_t)bh * Dd * Tt;
  const uint32_t kB = (uint32_t)(uintptr_t)&Ks[0][0][0];
  const uint32_t vB = (uint32_t)(uintptr_t)&Vs[0][0][0];

  const v16h aq0 = load_a_frag(Qp + (size_t)q0 * Dd + 0,  Dd);
  const v16h aq1 = load_a_frag(Qp + (size_t)q0 * Dd + 32, Dd);

  v8f o[4] = {};
  float mrun[8], lrun[8];
#pragma unroll
  for (int r = 0; r < 8; ++r) { mrun[r] = -3.0e38f; lrun[r] = 0.f; }

  stage_tile<3, 512>(kB, Kp, Dd);   // 64 x 64 key chunk 0
  stage_tile<3, 512>(vB, Vp, Tt);   // 64 x 64 value chunk 0 ([d][t])
  wait_async0();
  __syncthreads();

  for (int kc = 0; kc < Tt; kc += 64) {
    const int buf = (kc >> 6) & 1;
    if (kc + 64 < Tt) {
      stage_tile<3, 512>(kB + (uint32_t)(buf ^ 1) * 8192u,
                         Kp + (size_t)(kc + 64) * Dd, Dd);
      stage_tile<3, 512>(vB + (uint32_t)(buf ^ 1) * 8192u, Vp + kc + 64, Tt);
    }
    // S = Q K^T for this 16x64 key chunk: gather 8 fragments, 8 WMMAs
    const f16* Kc = &Ks[buf][0][0];
    v16h bk[8];
#pragma unroll
    for (int jt = 0; jt < 4; ++jt) {
      bk[2 * jt]     = load_b_frag(Kc + jt * 16 * 64, 64);
      bk[2 * jt + 1] = load_b_frag(Kc + jt * 16 * 64 + 32, 64);
    }
    v8f s[4];
#pragma unroll
    for (int jt = 0; jt < 4; ++jt) {
      v8f sj = {};
      sj = wmma16(aq0, bk[2 * jt], sj);
      sj = wmma16(aq1, bk[2 * jt + 1], sj);
      s[jt] = sj;
    }
    // online softmax; row M = r + 8*sel lives across one 16-lane half
    float fr[8];
#pragma unroll
    for (int r = 0; r < 8; ++r) {
      float v0 = s[0][r] * 0.125f, v1 = s[1][r] * 0.125f;   // 1/sqrt(64)
      float v2 = s[2][r] * 0.125f, v3 = s[3][r] * 0.125f;
      float mx = fmaxf(fmaxf(v0, v1), fmaxf(v2, v3));
#pragma unroll
      for (int off = 1; off < 16; off <<= 1)
        mx = fmaxf(mx, __shfl_xor(mx, off, 32));
      float mnew = fmaxf(mrun[r], mx);
      float p0 = __expf(v0 - mnew), p1 = __expf(v1 - mnew);
      float p2 = __expf(v2 - mnew), p3 = __expf(v3 - mnew);
      float rs = (p0 + p1) + (p2 + p3);
#pragma unroll
      for (int off = 1; off < 16; off <<= 1)
        rs += __shfl_xor(rs, off, 32);
      float f = __expf(mrun[r] - mnew);
      lrun[r] = lrun[r] * f + rs;
      mrun[r] = mnew;
      fr[r] = f;
      s[0][r] = p0; s[1][r] = p1; s[2][r] = p2; s[3][r] = p3;
    }
#pragma unroll
    for (int nt = 0; nt < 4; ++nt)
#pragma unroll
      for (int r = 0; r < 8; ++r) o[nt][r] *= fr[r];

    // prefetch V fragments (independent of the pbuf round-trip)
    v16h bv[8];
#pragma unroll
    for (int nt = 0; nt < 4; ++nt) {
      const f16* Vc = &Vs[buf][nt * 16][0];
      bv[2 * nt]     = load_b_frag(Vc, 64);
      bv[2 * nt + 1] = load_b_frag(Vc + 32, 64);
    }
    // D-layout probs -> LDS -> A-fragments (per-wave buffer, intra-wave only)
#pragma unroll
    for (int jt = 0; jt < 4; ++jt)
#pragma unroll
      for (int r = 0; r < 8; ++r)
        pbuf[wid][r + 8 * sel][(lane & 15) + 16 * jt] = (f16)s[jt][r];
    asm volatile("s_wait_dscnt 0x0" ::: "memory");
    v16h ap0 = load_a_frag(&pbuf[wid][0][0], 64);       // keys kc+0..31
    v16h ap1 = load_a_frag(&pbuf[wid][0][0] + 32, 64);  // keys kc+32..63

    // O += P * V : 8 back-to-back WMMAs
#pragma unroll
    for (int nt = 0; nt < 4; ++nt) {
      o[nt] = wmma16(ap0, bv[2 * nt], o[nt]);
      o[nt] = wmma16(ap1, bv[2 * nt + 1], o[nt]);
    }
    wait_async0();
    __syncthreads();
  }

  // normalize and store Y[B,T,C] (c = h*64 + d), f16
#pragma unroll
  for (int r = 0; r < 8; ++r) {
    float inv = 1.0f / lrun[r];
    int t = q0 + r + 8 * sel;
#pragma unroll
    for (int nt = 0; nt < 4; ++nt) {
      int d = nt * 16 + (lane & 15);
      Y[(((size_t)b * Tt + t) * Hh + h) * Dd + d] = (f16)(o[nt][r] * inv);
    }
  }
}

// ---------------------------------------------------------------------------
// Output projection: [4096 x 1024] * [1024 x 1024] -> fp32 d_out
__global__ __launch_bounds__(256) void k_proj(const f16* __restrict__ Y,
                                              const f16* __restrict__ WpT,
                                              float* __restrict__ Out) {
  __shared__ f16 As[2][128][64];
  __shared__ f16 Bs[2][128][64];
  const int wid  = threadIdx.x >> 5;
  const int lane = threadIdx.x & 31;
  const int sel  = lane >> 4;
  const int wm = wid & 3, wn = wid >> 2;
  const int m0 = blockIdx.y * 128 + wm * 32;
  const int n0 = blockIdx.x * 128 + wn * 64;

  v8f acc[2][4] = {};
  gemm_core(Y + (size_t)(blockIdx.y * 128) * Cc,
            WpT + (size_t)(blockIdx.x * 128) * Cc, As, Bs, wm, wn, acc);

#pragma unroll
  for (int mt = 0; mt < 2; ++mt)
#pragma unroll
    for (int nt = 0; nt < 4; ++nt)
#pragma unroll
      for (int r = 0; r < 8; ++r) {
        int row = m0 + mt * 16 + r + 8 * sel;
        int col = n0 + nt * 16 + (lane & 15);
        Out[(size_t)row * Cc + col] = acc[mt][nt][r];
      }
}

// ---------------------------------------------------------------------------
extern "C" void kernel_launch(void* const* d_in, const int* in_sizes, int n_in,
                              void* d_out, int out_size, void* d_ws, size_t ws_size,
                              hipStream_t stream) {
  const float* x      = (const float*)d_in[0];   // [B,T,C]
  const float* w_attn = (const float*)d_in[1];   // [C,3C]
  const float* w_proj = (const float*)d_in[2];   // [C,C]
  float* out = (float*)d_out;                    // [B,T,C] fp32

  char* ws = (char*)d_ws;
  f16* X16 = (f16*)(ws + (size_t)( 0u << 20));   // 8 MB  [4096,1024]
  f16* WaT = (f16*)(ws + (size_t)( 8u << 20));   // 6 MB  [3072,1024] N-major
  f16* WpT = (f16*)(ws + (size_t)(14u << 20));   // 2 MB  [1024,1024] N-major
  f16* Q   = (f16*)(ws + (size_t)(16u << 20));   // 8 MB  [B,H,T,D]
  f16* Kd  = (f16*)(ws + (size_t)(24u << 20));   // 8 MB  [B,H,T,D]
  f16* Vt  = (f16*)(ws + (size_t)(32u << 20));   // 8 MB  [B,H,D,T]
  f16* Yb  = (f16*)(ws + (size_t)(40u << 20));   // 8 MB  [B,T,C]

  k_cvt  <<<(Mm * Cc) / 256, 256, 0, stream>>>(x, X16, Mm * Cc);
  k_cvt_t<<<(Cc * NQKV) / 256, 256, 0, stream>>>(w_attn, WaT, Cc, NQKV);
  k_cvt_t<<<(Cc * Cc) / 256, 256, 0, stream>>>(w_proj, WpT, Cc, Cc);

  k_qkv <<<dim3(NQKV / 128, Mm / 128), 256, 0, stream>>>(X16, WaT, Q, Kd, Vt);
  k_attn<<<512, 256, 0, stream>>>(Q, Kd, Vt, Yb);
  k_proj<<<dim3(Cc / 128, Mm / 128), 256, 0, stream>>>(Yb, WpT, out);
}
